// EHROntologyModel_65987877535832
// MI455X (gfx1250) — compile-verified, matrix operand order checked by reading, not run on hardware
//
#include <hip/hip_runtime.h>
#include <hip/hip_bf16.h>
#include <math.h>

typedef __attribute__((ext_vector_type(16))) _Float16 v16h;
typedef __attribute__((ext_vector_type(8)))  _Float16 v8h;
typedef __attribute__((ext_vector_type(8)))  float    v8f;

// ---------------------------------------------------------------------------
// GEMM via CDNA5 WMMA:  C[M,N] = A[M,K] * B[K,N] (+bias[N])
//   A16: f16 row-major [M,K]   (produced once by upstream kernels)
//   Bt : f16 row-major [N,K]   (B transposed + converted once)
//   C  : fp32 row-major [M,N]
// One wave computes a 16x128 strip (8 accumulators). Per k-step: the A
// fragment is loaded once, then ALL 8 B fragments are preloaded into distinct
// registers before the 8 WMMAs issue, so the loads overlap the matrix ops
// instead of serializing on s_wait_loadcnt 0. Requires K%32==0, N%128==0.
// Fragment layout per CDNA5 ISA 7.12.2 (16-bit A 16x32; B symmetric; C/D 8 VGPR).
// ---------------------------------------------------------------------------
#define GEMM_NT 8

__global__ void __launch_bounds__(256)
gemm_wmma(const _Float16* __restrict__ A16, const _Float16* __restrict__ Bt,
          const float* __restrict__ bias, float* __restrict__ C,
          int M, int N, int K)
{
  const int wave = threadIdx.x >> 5;
  const int lane = threadIdx.x & 31;
  const int nGroups = N >> 7;                   // groups of 8 N-tiles (128 cols)
  const int tilesM  = (M + 15) >> 4;
  const int tile = blockIdx.x * 8 + wave;
  if (tile >= tilesM * nGroups) return;         // wave-uniform exit (EXEC stays full)
  const int tm = tile / nGroups;
  const int nbase = (tile % nGroups) * 128;
  const int idx = lane & 15, h = lane >> 4;

  const int m  = tm * 16 + idx;
  const int mm = m < M ? m : (M - 1);           // clamp loads, mask stores

  const _Float16* arow  = A16 + (size_t)mm * K;
  const _Float16* bbase = Bt + (size_t)(nbase + idx) * K;
  const size_t bstride = (size_t)16 * K;        // halfs between adjacent N-tiles

  v8f acc[GEMM_NT];
#pragma unroll
  for (int t = 0; t < GEMM_NT; ++t)
    acc[t] = (v8f){0.f, 0.f, 0.f, 0.f, 0.f, 0.f, 0.f, 0.f};

  for (int k0 = 0; k0 < K; k0 += 32) {
    // A fragment: lane holds K = k0 + {h*8 .. h*8+7} and {16+h*8 .. 16+h*8+7}
    const v8h a0 = *(const v8h*)(arow + k0 + h * 8);
    const v8h a1 = *(const v8h*)(arow + k0 + 16 + h * 8);
    __builtin_prefetch(arow + k0 + 32, 0, 1);   // global_prefetch next k-step
    v16h af;
#pragma unroll
    for (int e = 0; e < 8; ++e) { af[e] = a0[e]; af[e + 8] = a1[e]; }

    v16h bf[GEMM_NT];                           // preload ALL B fragments first
#pragma unroll
    for (int t = 0; t < GEMM_NT; ++t) {
      const _Float16* brow = bbase + (size_t)t * bstride;
      const v8h b0 = *(const v8h*)(brow + k0 + h * 8);
      const v8h b1 = *(const v8h*)(brow + k0 + 16 + h * 8);
#pragma unroll
      for (int e = 0; e < 8; ++e) { bf[t][e] = b0[e]; bf[t][e + 8] = b1[e]; }
    }
#pragma unroll
    for (int t = 0; t < GEMM_NT; ++t)
      acc[t] = __builtin_amdgcn_wmma_f32_16x16x32_f16(false, af, false, bf[t],
                                                      (short)0, acc[t], false, false);
  }

#pragma unroll
  for (int t = 0; t < GEMM_NT; ++t) {
    const int col = nbase + t * 16 + idx;
    const float bv = bias ? bias[col] : 0.f;
#pragma unroll
    for (int r = 0; r < 8; ++r) {               // C/D: VGPR r -> row h*8+r, col idx
      const int row = tm * 16 + h * 8 + r;
      if (row < M) C[(size_t)row * N + col] = acc[t][r] + bv;
    }
  }
}

// B[K,N] fp32 -> Bt[N,K] f16
__global__ void __launch_bounds__(256)
transpose_cvt(const float* __restrict__ B, _Float16* __restrict__ Bt, int K, int N)
{
  int tid = blockIdx.x * 256 + threadIdx.x;
  if (tid >= K * N) return;
  int k = tid / N, n = tid % N;
  Bt[(size_t)n * K + k] = (_Float16)B[tid];
}

// elementwise fp32 -> f16 copy
__global__ void __launch_bounds__(256)
cvt_f16(const float* __restrict__ in, _Float16* __restrict__ out, int total)
{
  int tid = blockIdx.x * 256 + threadIdx.x;
  if (tid < total) out[tid] = (_Float16)in[tid];
}

// per-(node,head) attention scores: dot(xp[i,h,:128], a_src/a_dst[h])
__global__ void __launch_bounds__(256)
att_scores(const float* __restrict__ xp, const float* __restrict__ a_src,
           const float* __restrict__ a_dst, float* __restrict__ as_,
           float* __restrict__ ad_, int Nn, int heads)
{
  int tid = blockIdx.x * 256 + threadIdx.x;
  if (tid >= Nn * heads) return;
  int i = tid / heads, h = tid % heads;
  const float* x = xp + (size_t)i * heads * 128 + h * 128;
  float s = 0.f, d = 0.f;
  for (int c = 0; c < 128; ++c) {
    s += x[c] * a_src[h * 128 + c];
    d += x[c] * a_dst[h * 128 + c];
  }
  as_[tid] = s;
  ad_[tid] = d;
}

// softmax numerators (edges + self loops) + segment-sum denominators
__global__ void __launch_bounds__(256)
edge_softmax_num(const int* __restrict__ src, const int* __restrict__ dst,
                 int E, int Nn, int heads,
                 const float* __restrict__ as_, const float* __restrict__ ad_,
                 float* __restrict__ ew, float* __restrict__ den)
{
  int tid = blockIdx.x * 256 + threadIdx.x;
  if (tid >= E + Nn) return;
  int s = tid < E ? src[tid] : (tid - E);
  int d = tid < E ? dst[tid] : (tid - E);
  for (int h = 0; h < heads; ++h) {
    float e = as_[s * heads + h] + ad_[d * heads + h];
    e = e > 0.f ? e : 0.2f * e;                 // leaky_relu(0.2)
    float ex = __expf(e);                       // values are tiny -> max-shift not needed
    ew[(size_t)tid * heads + h] = ex;
    atomicAdd(&den[d * heads + h], ex);
  }
}

// out[dst] += xp[src] * alpha  (block per edge, thread per channel)
__global__ void edge_scatter(const int* __restrict__ src, const int* __restrict__ dst,
                             int E, int heads,
                             const float* __restrict__ xp, const float* __restrict__ ew,
                             const float* __restrict__ den, float* __restrict__ out)
{
  int e = blockIdx.x;
  int c = threadIdx.x;                          // heads*128 threads
  int s = e < E ? src[e] : (e - E);
  int d = e < E ? dst[e] : (e - E);
  int h = c >> 7;
  int C = heads * 128;
  float alpha = ew[(size_t)e * heads + h] / (den[d * heads + h] + 1e-16f);
  atomicAdd(&out[(size_t)d * C + c], xp[(size_t)s * C + c] * alpha);
}

// x += bias (broadcast over rows); optional ELU; optional f16 mirror for WMMA-A
__global__ void __launch_bounds__(256)
bias_act(float* __restrict__ x, const float* __restrict__ b, int total, int C,
         int elu, _Float16* __restrict__ xh)
{
  int tid = blockIdx.x * 256 + threadIdx.x;
  if (tid >= total) return;
  float v = x[tid] + b[tid % C];
  if (elu) v = v > 0.f ? v : expm1f(v);
  x[tid] = v;
  if (xh) xh[tid] = (_Float16)v;
}

// all_emb[base+j] = x[map[j]]   (128 channels)
__global__ void gather_rows(const float* __restrict__ xsrc, const int* __restrict__ map,
                            float* __restrict__ dstbuf)
{
  int j = blockIdx.x, c = threadIdx.x;          // 128 threads
  dstbuf[(size_t)j * 128 + c] = xsrc[(size_t)map[j] * 128 + c];
}

// x0[i] = concat(all_emb[x_idx[i]], text_proj[x_idx[i]]) ; f32 + f16 mirror
__global__ void __launch_bounds__(256)
build_x0(const int* __restrict__ xidx, const float* __restrict__ all_emb,
         const float* __restrict__ text_proj, float* __restrict__ x0,
         _Float16* __restrict__ x0h)
{
  int i = blockIdx.x, c = threadIdx.x;          // 256 threads
  int v = xidx[i];
  float val = (c < 128) ? all_emb[(size_t)v * 128 + c]
                        : text_proj[(size_t)v * 128 + (c - 128)];
  x0[(size_t)i * 256 + c] = val;
  x0h[(size_t)i * 256 + c] = (_Float16)val;
}

// NodeFilter + global_add_pool
__global__ void __launch_bounds__(256)
filter_pool(const float* __restrict__ node, const float* __restrict__ x0,
            const float* __restrict__ filt, const int* __restrict__ batch,
            float* __restrict__ pooled)
{
  __shared__ float red[256];
  int i = blockIdx.x, c = threadIdx.x;
  float nv = node[(size_t)i * 256 + c];
  float xv = x0[(size_t)i * 256 + c];
  red[c] = nv * filt[c] * xv;
  __syncthreads();
  for (int k = 128; k > 0; k >>= 1) {
    if (c < k) red[c] += red[c + k];
    __syncthreads();
  }
  float s = 1.f / (1.f + __expf(-red[0]));
  atomicAdd(&pooled[(size_t)batch[i] * 256 + c], s * (nv + xv));
}

// ms[b,t] = sum_e Z[b, e*16+t] * gr[b,e]   (LDS f32 atomics)
__global__ void __launch_bounds__(256)
ntm_ms(const float* __restrict__ Z, const float* __restrict__ gr, float* __restrict__ ms)
{
  __shared__ float acc[16];
  int b = blockIdx.x, e = threadIdx.x;          // 256 threads
  if (e < 16) acc[e] = 0.f;
  __syncthreads();
  float g = gr[(size_t)b * 256 + e];
  const float* z = Z + (size_t)b * 4096 + e * 16;
#pragma unroll
  for (int t = 0; t < 16; ++t) atomicAdd(&acc[t], g * z[t]);   // ds_add_f32
  __syncthreads();
  if (e < 16) ms[b * 16 + e] = acc[e];
}

// pair = tanh(ms + [gl,gr]@Wb + b)
__global__ void __launch_bounds__(256)
ntm_pair(const float* __restrict__ gl, const float* __restrict__ gr,
         const float* __restrict__ Wb, const float* __restrict__ ms,
         const float* __restrict__ nb, float* __restrict__ pair)
{
  int tid = blockIdx.x * 256 + threadIdx.x;
  if (tid >= 1024 * 16) return;
  int b = tid >> 4, t = tid & 15;
  const float* l = gl + (size_t)b * 256;
  const float* r = gr + (size_t)b * 256;
  float a = 0.f;
  for (int j = 0; j < 256; ++j) a += l[j] * Wb[j * 16 + t];
  for (int j = 0; j < 256; ++j) a += r[j] * Wb[(256 + j) * 16 + t];
  pair[tid] = tanhf(ms[tid] + a + nb[t]);
}

// logits + per-row NLL
__global__ void __launch_bounds__(256)
cls_loss(const float* __restrict__ pair, const float* __restrict__ W,
         const float* __restrict__ bc, const int* __restrict__ label,
         float* __restrict__ logits_out, float* __restrict__ lossbuf)
{
  int b = blockIdx.x * 256 + threadIdx.x;
  if (b >= 1024) return;
  float l0 = bc[0], l1 = bc[1];
  for (int t = 0; t < 16; ++t) {
    float p = pair[b * 16 + t];
    l0 += p * W[t * 2];
    l1 += p * W[t * 2 + 1];
  }
  logits_out[b * 2] = l0;
  logits_out[b * 2 + 1] = l1;
  float m = fmaxf(l0, l1);
  float lse = m + logf(__expf(l0 - m) + __expf(l1 - m));
  lossbuf[b] = lse - (label[b] == 0 ? l0 : l1);
}

__global__ void __launch_bounds__(256)
loss_reduce(const float* __restrict__ lossbuf, float* __restrict__ out)
{
  __shared__ float sm[256];
  int t = threadIdx.x;
  float s = 0.f;
  for (int i = t; i < 1024; i += 256) s += lossbuf[i];
  sm[t] = s;
  __syncthreads();
  for (int k = 128; k > 0; k >>= 1) {
    if (t < k) sm[t] += sm[t + k];
    __syncthreads();
  }
  if (t == 0) out[0] = sm[0] * (1.0f / 1024.0f);
}

// ---------------------------------------------------------------------------
extern "C" void kernel_launch(void* const* d_in, const int* in_sizes, int n_in,
                              void* d_out_, int out_size, void* d_ws, size_t ws_size,
                              hipStream_t stream)
{
  (void)in_sizes; (void)n_in; (void)out_size; (void)ws_size;
  // ---- input mapping: params/consts pytrees flattened in sorted-key order ----
  const float* cls_W    = (const float*)d_in[0];
  const float* cls_b    = (const float*)d_in[1];
  const float* filt_w   = (const float*)d_in[2];
  const float* g1_W     = (const float*)d_in[3];
  const float* g1_adst  = (const float*)d_in[4];
  const float* g1_asrc  = (const float*)d_in[5];
  const float* g1_b     = (const float*)d_in[6];
  const float* g2_W     = (const float*)d_in[7];
  const float* g2_adst  = (const float*)d_in[8];
  const float* g2_asrc  = (const float*)d_in[9];
  const float* g2_b     = (const float*)d_in[10];
  const float* ntm_W    = (const float*)d_in[11];   // [256,256,16] == [256,4096]
  const float* ntm_Wb   = (const float*)d_in[12];   // [512,16]
  const float* ntm_b    = (const float*)d_in[13];
  // onto (alphabetical): atc=14..22, diag=23..31, proce=32..40; each: emb,g1W,g1adst,g1asrc,g1b,g2W,g2adst,g2asrc,g2b
  const float* text_W   = (const float*)d_in[41];   // [768,128]
  const float* text_b   = (const float*)d_in[42];
  const float* text_emb = (const float*)d_in[43];   // [8300,768]
  // consts (alphabetical): atc_e1=44,atc_e2=45,atc_map=46, diag=47..49, proce=50..52
  const int* left_x     = (const int*)d_in[53];
  const int* left_gi    = (const int*)d_in[54];
  const int* left_batch = (const int*)d_in[55];
  const int* right_x    = (const int*)d_in[56];
  const int* right_gi   = (const int*)d_in[57];
  const int* right_batch= (const int*)d_in[58];
  const int* label      = (const int*)d_in[59];

  float* d_out = (float*)d_out_;                    // [0]=loss, [1..2048]=logits

  const int NB = 65536, EB = 262144;

  // ---- workspace bump allocator ----
  unsigned char* wsb = (unsigned char*)d_ws;
  size_t off = 0;
  auto alloc = [&](size_t bytes) -> void* {
    void* p = wsb + off;
    off += (bytes + 255) & ~(size_t)255;
    return p;
  };
  _Float16* wt_onto = (_Float16*)alloc(6 * 128 * 128 * sizeof(_Float16));
  _Float16* wt_text = (_Float16*)alloc(128 * 768 * sizeof(_Float16));
  _Float16* wt_g1   = (_Float16*)alloc(256 * 256 * sizeof(_Float16));
  _Float16* wt_g2   = (_Float16*)alloc(256 * 256 * sizeof(_Float16));
  _Float16* wt_ntm  = (_Float16*)alloc((size_t)4096 * 256 * sizeof(_Float16));
  _Float16* emb16     = (_Float16*)alloc((size_t)12000 * 128 * sizeof(_Float16));
  _Float16* ox1h      = (_Float16*)alloc((size_t)12000 * 128 * sizeof(_Float16));
  _Float16* text_emb16= (_Float16*)alloc((size_t)8300 * 768 * sizeof(_Float16));
  _Float16* x0h       = (_Float16*)alloc((size_t)NB * 256 * sizeof(_Float16));
  _Float16* X1h       = (_Float16*)alloc((size_t)NB * 256 * sizeof(_Float16));
  _Float16* gl16      = (_Float16*)alloc((size_t)1024 * 256 * sizeof(_Float16));
  float* oxp  = (float*)alloc((size_t)12000 * 128 * sizeof(float));
  float* ox1  = (float*)alloc((size_t)12000 * 128 * sizeof(float));
  float* oas  = (float*)alloc(12000 * sizeof(float));
  float* oad  = (float*)alloc(12000 * sizeof(float));
  float* oden = (float*)alloc(12000 * sizeof(float));
  float* oew  = (float*)alloc(36000 * sizeof(float));
  float* all_emb   = (float*)alloc((size_t)8300 * 128 * sizeof(float));
  float* text_proj = (float*)alloc((size_t)8300 * 128 * sizeof(float));
  float* x0  = (float*)alloc((size_t)NB * 256 * sizeof(float));
  float* XP  = (float*)alloc((size_t)NB * 256 * sizeof(float));
  float* X1  = (float*)alloc((size_t)NB * 256 * sizeof(float));   // also "node"
  float* as2 = (float*)alloc((size_t)NB * 2 * sizeof(float));
  float* ad2 = (float*)alloc((size_t)NB * 2 * sizeof(float));
  float* den2= (float*)alloc((size_t)NB * 2 * sizeof(float));
  float* ew2 = (float*)alloc((size_t)(EB + NB) * 2 * sizeof(float));
  float* gl  = (float*)alloc((size_t)1024 * 256 * sizeof(float));
  float* gr  = (float*)alloc((size_t)1024 * 256 * sizeof(float));
  float* Zb  = (float*)alloc((size_t)1024 * 4096 * sizeof(float));
  float* ms  = (float*)alloc(1024 * 16 * sizeof(float));
  float* pairb = (float*)alloc(1024 * 16 * sizeof(float));
  float* lossbuf = (float*)alloc(1024 * sizeof(float));

  auto gemm = [&](const _Float16* A16, const _Float16* Bt, const float* bias, float* C,
                  int M, int N, int K) {
    int tiles = ((M + 15) / 16) * (N / 128);      // one wave per 16x128 strip
    gemm_wmma<<<(tiles + 7) / 8, 256, 0, stream>>>(A16, Bt, bias, C, M, N, K);
  };
  auto tcvt = [&](const float* B, _Float16* Bt, int K, int N) {
    transpose_cvt<<<(K * N + 255) / 256, 256, 0, stream>>>(B, Bt, K, N);
  };
  auto cvt = [&](const float* in, _Float16* out, int total) {
    cvt_f16<<<(total + 255) / 256, 256, 0, stream>>>(in, out, total);
  };

  // ---- weight transpose/convert + f16 activations (once per launch) ----
  tcvt(text_W, wt_text, 768, 128);
  tcvt(g1_W, wt_g1, 256, 256);
  tcvt(g2_W, wt_g2, 256, 256);
  tcvt(ntm_W, wt_ntm, 256, 4096);
  cvt(text_emb, text_emb16, 8300 * 768);

  // ---- ontology GATs: process diag(base 0), proce(6000), atc(8000) ----
  struct OntoG { int pbase, cbase, n, e, voc, obase; };
  const OntoG og[3] = { {23, 47, 12000, 24000, 6000, 0},
                        {32, 50,  4000,  8000, 2000, 6000},
                        {14, 44,   600,  1200,  300, 8000} };
  for (int g = 0; g < 3; ++g) {
    const float* emb  = (const float*)d_in[og[g].pbase + 0];
    const float* o1W  = (const float*)d_in[og[g].pbase + 1];
    const float* o1ad = (const float*)d_in[og[g].pbase + 2];
    const float* o1as = (const float*)d_in[og[g].pbase + 3];
    const float* o1b  = (const float*)d_in[og[g].pbase + 4];
    const float* o2W  = (const float*)d_in[og[g].pbase + 5];
    const float* o2ad = (const float*)d_in[og[g].pbase + 6];
    const float* o2as = (const float*)d_in[og[g].pbase + 7];
    const float* o2b  = (const float*)d_in[og[g].pbase + 8];
    const int* e1 = (const int*)d_in[og[g].cbase + 0];
    const int* e2 = (const int*)d_in[og[g].cbase + 1];
    const int* map= (const int*)d_in[og[g].cbase + 2];
    int n = og[g].n, e = og[g].e;
    _Float16* wt1 = wt_onto + (size_t)g * 2 * 16384;
    _Float16* wt2 = wt1 + 16384;
    tcvt(o1W, wt1, 128, 128);
    tcvt(o2W, wt2, 128, 128);
    cvt(emb, emb16, n * 128);

    const _Float16* lin[2] = {emb16, ox1h};
    const _Float16* lwt[2] = {wt1, wt2};
    const float* las[2] = {o1as, o2as};
    const float* lad[2] = {o1ad, o2ad};
    const float* lb[2]  = {o1b, o2b};
    const int* le[2] = {e1, e2};
    for (int layer = 0; layer < 2; ++layer) {
      gemm(lin[layer], lwt[layer], nullptr, oxp, n, 128, 128);
      att_scores<<<(n + 255) / 256, 256, 0, stream>>>(oxp, las[layer], lad[layer], oas, oad, n, 1);
      hipMemsetAsync(oden, 0, n * sizeof(float), stream);
      hipMemsetAsync(ox1, 0, (size_t)n * 128 * sizeof(float), stream);
      edge_softmax_num<<<(e + n + 255) / 256, 256, 0, stream>>>(
          le[layer], le[layer] + e, e, n, 1, oas, oad, oew, oden);
      edge_scatter<<<e + n, 128, 0, stream>>>(le[layer], le[layer] + e, e, 1, oxp, oew, oden, ox1);
      bias_act<<<((size_t)n * 128 + 255) / 256, 256, 0, stream>>>(
          ox1, lb[layer], n * 128, 128, 0, layer == 0 ? ox1h : (_Float16*)nullptr);
    }
    gather_rows<<<og[g].voc, 128, 0, stream>>>(ox1, map, all_emb + (size_t)og[g].obase * 128);
  }

  // ---- vocab-level text projection (algebraic hoist of the per-node GEMM) ----
  gemm(text_emb16, wt_text, text_b, text_proj, 8300, 128, 768);

  // ---- encoder per side ----
  const int* sx[2] = {left_x, right_x};
  const int* sg[2] = {left_gi, right_gi};
  const int* sb[2] = {left_batch, right_batch};
  float* pooled[2] = {gl, gr};
  for (int s = 0; s < 2; ++s) {
    build_x0<<<NB, 256, 0, stream>>>(sx[s], all_emb, text_proj, x0, x0h);
    const _Float16* lin[2] = {x0h, X1h};
    const _Float16* lwt[2] = {wt_g1, wt_g2};
    const float* las[2] = {g1_asrc, g2_asrc};
    const float* lad[2] = {g1_adst, g2_adst};
    const float* lb[2]  = {g1_b, g2_b};
    for (int layer = 0; layer < 2; ++layer) {
      gemm(lin[layer], lwt[layer], nullptr, XP, NB, 256, 256);
      att_scores<<<(NB * 2 + 255) / 256, 256, 0, stream>>>(XP, las[layer], lad[layer], as2, ad2, NB, 2);
      hipMemsetAsync(den2, 0, (size_t)NB * 2 * sizeof(float), stream);
      hipMemsetAsync(X1, 0, (size_t)NB * 256 * sizeof(float), stream);
      edge_softmax_num<<<(EB + NB + 255) / 256, 256, 0, stream>>>(
          sg[s], sg[s] + EB, EB, NB, 2, as2, ad2, ew2, den2);
      edge_scatter<<<EB + NB, 256, 0, stream>>>(sg[s], sg[s] + EB, EB, 2, XP, ew2, den2, X1);
      bias_act<<<((size_t)NB * 256 + 255) / 256, 256, 0, stream>>>(
          X1, lb[layer], NB * 256, 256, layer == 0 ? 1 : 0,
          layer == 0 ? X1h : (_Float16*)nullptr);
    }
    hipMemsetAsync(pooled[s], 0, (size_t)1024 * 256 * sizeof(float), stream);
    filter_pool<<<NB, 256, 0, stream>>>(X1, x0, filt_w, sb[s], pooled[s]);
  }

  // ---- neural tensor module + classifier + loss ----
  cvt(gl, gl16, 1024 * 256);
  gemm(gl16, wt_ntm, nullptr, Zb, 1024, 4096, 256);
  ntm_ms<<<1024, 256, 0, stream>>>(Zb, gr, ms);
  ntm_pair<<<(1024 * 16 + 255) / 256, 256, 0, stream>>>(gl, gr, ntm_Wb, ms, ntm_b, pairb);
  cls_loss<<<4, 256, 0, stream>>>(pairb, cls_W, cls_b, label, d_out + 1, lossbuf);
  loss_reduce<<<1, 256, 0, stream>>>(lossbuf, d_out);
}
